// PerspectiveEventHead_75952201662949
// MI455X (gfx1250) — matrix operand order repaired and checked by reading, not compiled, not run on hardware
//
#include <hip/hip_runtime.h>

// ---------------------------------------------------------------------------
// Sizes from the reference
// ---------------------------------------------------------------------------
#define NN 256   // N personalities
#define PP 256   // P
#define DD 512   // D
#define HH 128   // H
#define BB 256   // batch
#define NBLK 3

#define PA 264   // padded LDS pitch in halves (264*2B = 528B -> 4-bank row skew)

typedef __attribute__((ext_vector_type(16))) _Float16 v16h;
typedef __attribute__((ext_vector_type(8)))  _Float16 v8h;
typedef __attribute__((ext_vector_type(8)))  float    v8f;

// ---------------------------------------------------------------------------
// Kernel 1: abd_p[b][p] = abd[b] . align_w[p] + align_b[p]
// ---------------------------------------------------------------------------
__global__ void peh_align_kernel(const float* __restrict__ abd,
                                 const float* __restrict__ alw,
                                 const float* __restrict__ alb,
                                 float* __restrict__ abd_p) {
  const int b = blockIdx.x;
  const int p = threadIdx.x;
  const float* a = abd + (size_t)b * DD;
  const float* w = alw + (size_t)p * DD;
  float s = alb[p];
#pragma unroll 8
  for (int d = 0; d < DD; ++d) s += a[d] * w[d];
  abd_p[(size_t)b * PP + p] = s;
}

// ---------------------------------------------------------------------------
// Kernel 2: per-n mini-model bank.
//   grid.x = N*2 (n = bx>>1, batch half = bx&1), block = 256 threads (8 waves)
//   LDS: Ws[256][PA] f16  (W1_w[n], row-major q x p)
//        As[128][PA] f16  (softmax(256*C[b,n,:]))
//        Cs[128][PA] f16  (c_star)
//   GEMM 128x256x256 via v_wmma_f32_16x16x32_f16, then Wp-softmax/tanh/poll.
// ---------------------------------------------------------------------------
__global__ void __launch_bounds__(256)
peh_bank_kernel(const float* __restrict__ C,
                const float* __restrict__ W1w,
                const float* __restrict__ W1b,
                const float* __restrict__ Wpw,
                const float* __restrict__ Wpb,
                const float* __restrict__ abd_p,
                float* __restrict__ v_out) {
  extern __shared__ _Float16 smem[];
  _Float16* Ws = smem;               // 256*PA halves
  _Float16* As = Ws + 256 * PA;      // 128*PA halves
  _Float16* Cs = As + 128 * PA;      // 128*PA halves

  const int n   = blockIdx.x >> 1;
  const int b0  = (blockIdx.x & 1) * 128;
  const int tid = threadIdx.x;

  // ---------------- Phase A: stage + convert operands into LDS -------------
  if (tid < 128) {
    // waves 0-3: softmax(256 * C[b0+tid, n, :]) -> As (f16)
    const int r = tid;
    const float* crow = C + ((size_t)(b0 + r) * NN + n) * PP;
    float m = -1e30f;
#pragma unroll 4
    for (int p = 0; p < PP; ++p) m = fmaxf(m, crow[p]);
    float s = 0.0f;
    _Float16* arow = As + r * PA;
#pragma unroll 4
    for (int p = 0; p < PP; ++p) {
      float e = __expf(256.0f * crow[p] - 256.0f * m);
      s += e;
      arow[p] = (_Float16)e;
    }
    float inv = 1.0f / s;
#pragma unroll 4
    for (int p = 0; p < PP; ++p)
      arow[p] = (_Float16)((float)arow[p] * inv);
  } else {
    // waves 4-7: W1_w[n] f32 -> f16 into Ws (row-major [q][p])
    const int idx = tid - 128;
    const float* wsrc = W1w + (size_t)n * PP * PP;
#pragma unroll 4
    for (int e = idx; e < PP * PP; e += 128) {
      const int q = e >> 8;
      const int p = e & 255;
      Ws[q * PA + p] = (_Float16)wsrc[e];
    }
  }
  __syncthreads();

  // ---------------- Phase B: 128x256x256 GEMM with WMMA --------------------
  const int wave = tid >> 5;
  const int lane = tid & 31;
  const int lm = lane & 15;   // row-in-tile / col-in-tile
  const int lh = lane >> 4;   // K-half selector
  const int RM = (wave & 3) * 32;   // 4 waves tile M=128
  const int WN = (wave >> 2) * 64;  // 2 waves tile half of N per pass

#pragma unroll 1
  for (int pass = 0; pass < 2; ++pass) {
    const int Noff = pass * 128 + WN;
    v8f acc[2][4];
#pragma unroll
    for (int mi = 0; mi < 2; ++mi)
#pragma unroll
      for (int ni = 0; ni < 4; ++ni) acc[mi][ni] = (v8f){};

#pragma unroll 2
    for (int kt = 0; kt < 8; ++kt) {
      const int k0 = kt * 32;
      v16h afr[2], bfr[4];
#pragma unroll
      for (int mi = 0; mi < 2; ++mi) {
        // A 16x32 f16 layout: lane(l<16): K=0..7 & 16..23 ; lane(l>=16): +8
        const _Float16* ap = As + (RM + mi * 16 + lm) * PA + k0 + lh * 8;
        v8h lo = *(const v8h*)ap;
        v8h hi = *(const v8h*)(ap + 16);
        afr[mi] = __builtin_shufflevector(lo, hi, 0, 1, 2, 3, 4, 5, 6, 7,
                                          8, 9, 10, 11, 12, 13, 14, 15);
      }
#pragma unroll
      for (int ni = 0; ni < 4; ++ni) {
        // B 32x16 f16 layout: lane col = lm, K = 8*lh..8*lh+7 & 16+8*lh..16+8*lh+7
        const _Float16* bp = Ws + (Noff + ni * 16 + lm) * PA + k0 + lh * 8;
        v8h lo = *(const v8h*)bp;
        v8h hi = *(const v8h*)(bp + 16);
        bfr[ni] = __builtin_shufflevector(lo, hi, 0, 1, 2, 3, 4, 5, 6, 7,
                                          8, 9, 10, 11, 12, 13, 14, 15);
      }
#pragma unroll
      for (int mi = 0; mi < 2; ++mi)
#pragma unroll
        for (int ni = 0; ni < 4; ++ni)
          acc[mi][ni] = __builtin_amdgcn_wmma_f32_16x16x32_f16(
              false, afr[mi], false, bfr[ni], (short)0, acc[mi][ni],
              false, false);
    }

    // epilogue: bias + ReLU -> Cs (f16).  c[v] maps to M = v + 8*lh, N = lm.
#pragma unroll
    for (int ni = 0; ni < 4; ++ni) {
      const int q = Noff + ni * 16 + lm;
      const float bq = W1b[n * PP + q];
#pragma unroll
      for (int mi = 0; mi < 2; ++mi) {
#pragma unroll
        for (int v = 0; v < 8; ++v) {
          const int row = RM + mi * 16 + lh * 8 + v;
          float cs = acc[mi][ni][v] + bq;
          cs = fmaxf(cs, 0.0f);
          Cs[row * PA + q] = (_Float16)cs;
        }
      }
    }
  }
  __syncthreads();

  // ---------------- Phase C: Wp softmax, tanh mix, poll --------------------
  // two threads per batch row (adjacent lanes), split the q range
  const int r  = tid >> 1;
  const int qh = tid & 1;
  const int b  = b0 + r;
  const _Float16* csrow = Cs + r * PA + qh * 128;
  const float* wp0 = Wpw + (size_t)n * 2 * PP + qh * 128;
  const float* wp1 = wp0 + PP;
  float d0 = 0.0f, d1 = 0.0f;
#pragma unroll 4
  for (int q = 0; q < 128; ++q) {
    const float cs = (float)csrow[q];
    d0 += wp0[q] * cs;
    d1 += wp1[q] * cs;
  }
  d0 += __shfl_xor(d0, 1);
  d1 += __shfl_xor(d1, 1);
  d0 += Wpb[n * 2 + 0];
  d1 += Wpb[n * 2 + 1];
  const float mm = fmaxf(d0, d1);
  const float e0 = __expf(d0 - mm);
  const float e1 = __expf(d1 - mm);
  const float pin = 1.0f / (e0 + e1);
  const float p0 = e0 * pin, p1 = e1 * pin;

  const float* ap = abd_p + (size_t)b * PP + qh * 128;
  float vv = 0.0f;
#pragma unroll 4
  for (int q = 0; q < 128; ++q) {
    const float cs = (float)csrow[q];
    vv += (tanhf(p0 * cs) + tanhf(p1 * cs)) * ap[q];
  }
  vv += __shfl_xor(vv, 1);
  if (qh == 0) v_out[(size_t)b * NN + n] = vv;
}

// ---------------------------------------------------------------------------
// Bottleneck blocks (tiny GEMVs) + head
// ---------------------------------------------------------------------------
__global__ void peh_bn_up_kernel(const float* __restrict__ v,
                                 const float* __restrict__ w,
                                 const float* __restrict__ bias,
                                 float* __restrict__ h) {
  const int b = blockIdx.x, j = threadIdx.x;  // 128 threads
  const float* vr = v + (size_t)b * NN;
  const float* wr = w + (size_t)j * NN;
  float s = bias[j];
#pragma unroll 8
  for (int i = 0; i < NN; ++i) s += wr[i] * vr[i];
  h[(size_t)b * HH + j] = fmaxf(s, 0.0f);
}

__global__ void peh_bn_dn_kernel(const float* __restrict__ h,
                                 const float* __restrict__ w,
                                 const float* __restrict__ bias,
                                 float* __restrict__ v) {
  const int b = blockIdx.x, i = threadIdx.x;  // 256 threads
  const float* hr = h + (size_t)b * HH;
  const float* wr = w + (size_t)i * HH;
  float s = bias[i];
#pragma unroll 8
  for (int j = 0; j < HH; ++j) s += wr[j] * hr[j];
  v[(size_t)b * NN + i] = s;
}

__global__ void peh_head_kernel(const float* __restrict__ v,
                                const float* __restrict__ Ww,
                                const float* __restrict__ Wb,
                                float* __restrict__ out) {
  const int b = threadIdx.x;  // 256 threads, 1 block
  const float* vr = v + (size_t)b * NN;
  float s0 = Wb[0], s1 = Wb[1];
#pragma unroll 8
  for (int i = 0; i < NN; ++i) {
    s0 += Ww[i] * vr[i];
    s1 += Ww[NN + i] * vr[i];
  }
  const float m = fmaxf(s0, s1);
  const float e0 = __expf(s0 - m), e1 = __expf(s1 - m);
  const float inv = 1.0f / (e0 + e1);
  out[b * 2 + 0] = e0 * inv;
  out[b * 2 + 1] = e1 * inv;
}

// ---------------------------------------------------------------------------
// Launcher
// ---------------------------------------------------------------------------
extern "C" void kernel_launch(void* const* d_in, const int* in_sizes, int n_in,
                              void* d_out, int out_size, void* d_ws,
                              size_t ws_size, hipStream_t stream) {
  const float* C    = (const float*)d_in[0];
  const float* abd  = (const float*)d_in[1];
  const float* W1w  = (const float*)d_in[2];
  const float* W1b  = (const float*)d_in[3];
  const float* Wpw  = (const float*)d_in[4];
  const float* Wpb  = (const float*)d_in[5];
  const float* alw  = (const float*)d_in[6];
  const float* alb  = (const float*)d_in[7];
  const float* buw  = (const float*)d_in[8];
  const float* bub  = (const float*)d_in[9];
  const float* bdw  = (const float*)d_in[10];
  const float* bdb  = (const float*)d_in[11];
  const float* Wow  = (const float*)d_in[12];
  const float* Wob  = (const float*)d_in[13];

  float* out   = (float*)d_out;
  float* ws    = (float*)d_ws;
  float* abd_p = ws;                       // 65536 floats
  float* vbuf  = ws + (size_t)BB * PP;     // 65536 floats
  float* hbuf  = vbuf + (size_t)BB * NN;   // 32768 floats

  peh_align_kernel<<<BB, PP, 0, stream>>>(abd, alw, alb, abd_p);

  const size_t smem_bytes = (size_t)(256 + 128 + 128) * PA * 2;  // ~270 KB LDS
  peh_bank_kernel<<<NN * 2, 256, smem_bytes, stream>>>(C, W1w, W1b, Wpw, Wpb,
                                                       abd_p, vbuf);

  for (int k = 0; k < NBLK; ++k) {
    peh_bn_up_kernel<<<BB, HH, 0, stream>>>(vbuf, buw + (size_t)k * HH * NN,
                                            bub + (size_t)k * HH, hbuf);
    peh_bn_dn_kernel<<<BB, NN, 0, stream>>>(hbuf, bdw + (size_t)k * NN * HH,
                                            bdb + (size_t)k * NN, vbuf);
  }

  peh_head_kernel<<<1, BB, 0, stream>>>(vbuf, Wow, Wob, out);
}
